// NNConv_pair_42073499632117
// MI455X (gfx1250) — matrix operand order repaired and checked by reading, not compiled, not run on hardware
//
#include <hip/hip_runtime.h>

typedef __attribute__((ext_vector_type(16))) _Float16 v16h;
typedef __attribute__((ext_vector_type(8)))  float    v8f;

#define HD 16              // edge-MLP hidden width
#define FV 32              // F_IN == EMB == 32
#define FE 8               // edge attr width
#define KR (HD * FV + FV)  // 544 combined-weight rows (w2 rows + bias rows)

// ---------------- zero fill ----------------
__global__ void zero_kernel(float* __restrict__ p, int n) {
    int t = blockIdx.x * blockDim.x + threadIdx.x;
    if (t < n) p[t] = 0.0f;
}

// ---------------- edge MLP: h = relu(ea @ w1 + b1), [E,8]->[E,16] ----------------
__global__ void edge_mlp_kernel(const float* __restrict__ ea,
                                const float* __restrict__ w1,
                                const float* __restrict__ b1,
                                float* __restrict__ h, int E) {
    int t = blockIdx.x * blockDim.x + threadIdx.x;
    if (t >= E * HD) return;
    int e = t >> 4, k = t & 15;
    float acc = b1[k];
    #pragma unroll
    for (int j = 0; j < FE; ++j) acc += ea[e * FE + j] * w1[j * HD + k];
    h[t] = fmaxf(acc, 0.0f);
}

// ---------------- pack w2(+b2) -> f16, column-major [o][r], r in [0,544) ----------------
__global__ void prep_w2_kernel(const float* __restrict__ w2,   // [16,1024] == [512,32]
                               const float* __restrict__ b2,   // [1024] == [32,32]
                               _Float16* __restrict__ w2t) {
    int t = blockIdx.x * blockDim.x + threadIdx.x;
    if (t >= FV * KR) return;
    int o = t / KR, r = t % KR;
    float v = (r < HD * FV) ? w2[r * FV + o] : b2[(r - HD * FV) * FV + o];
    w2t[t] = (_Float16)v;
}

// ---------------- fused NNConv message GEMM via WMMA + scatter-add ----------------
// Per wave: a 16-edge tile.  msg[E,32] = U[E,544] @ W'[544,32] where
// U[e, 32*k + i] = h[e,k]*x[src[e], i] (k<16) and U[e, 512+i] = x[src[e], i].
// A fragments built in registers from the outer product; B streamed from w2t.
__global__ void msg_wmma_kernel(const float* __restrict__ xin,   // [N,32]
                                const int*   __restrict__ src,   // [E]
                                const int*   __restrict__ dst,   // [E]
                                const float* __restrict__ hbuf,  // [E,16]
                                const _Float16* __restrict__ w2t,// [32][544] col-major
                                float* __restrict__ agg,         // [N,32] (pre-zeroed)
                                int E, int ntiles) {
    int gtid = blockIdx.x * blockDim.x + threadIdx.x;
    int tile = gtid >> 5;
    int lane = threadIdx.x & 31;
    if (tile >= ntiles) return;     // wave-uniform exit

    const int  ebase = tile << 4;
    const int  m     = lane & 15;
    const bool hi    = (lane >= 16);
    const int  e     = ebase + m;
    const int  ec    = (e < E) ? e : (E - 1);
    const float mask = (e < E) ? 1.0f : 0.0f;

    const int s      = src[ec];
    const int kbaseA = hi ? 16 : 0;   // A: lanes>=16 hold K=16..31
    const int kbaseB = hi ? 16 : 0;   // B: lanes>=16 hold K=16..31

    // x row slice (k = kbaseA + j) loaded in A-fragment element order (compile-time offsets only)
    float xl[16];
    #pragma unroll
    for (int j = 0; j < 16; ++j) {
        const int kt = j;                       // K-within-tile minus kbaseA
        xl[j] = xin[s * FV + kbaseA + kt];
    }
    float hr[HD];
    #pragma unroll
    for (int k = 0; k < HD; ++k) hr[k] = mask * hbuf[ec * HD + k];

    v8f c0 = {}; v8f c1 = {};
    const _Float16* wcol0 = w2t + (size_t)m * KR + kbaseB;          // out cols 0..15
    const _Float16* wcol1 = w2t + (size_t)(m + 16) * KR + kbaseB;   // out cols 16..31

    #pragma unroll
    for (int kk = 0; kk < HD + 1; ++kk) {
        const float hk = (kk < HD) ? hr[kk] : mask;   // last step = bias rows (A = x)
        v16h a;
        #pragma unroll
        for (int j = 0; j < 16; ++j) a[j] = (_Float16)(hk * xl[j]);
        v16h b0 = *(const v16h*)(wcol0 + kk * FV);    // 32B-aligned contiguous K-slice
        v16h b1 = *(const v16h*)(wcol1 + kk * FV);
        c0 = __builtin_amdgcn_wmma_f32_16x16x32_f16(false, a, false, b0,
                                                    (short)0, c0, false, false);
        c1 = __builtin_amdgcn_wmma_f32_16x16x32_f16(false, a, false, b1,
                                                    (short)0, c1, false, false);
    }

    // D layout: VGPR r -> row M=r (+8 for lanes>=16), lane -> col n (=m / m+16)
    const int mrow = hi ? 8 : 0;
    #pragma unroll
    for (int r = 0; r < 8; ++r) {
        const int er = ebase + mrow + r;
        if (er < E) {
            const int dn = dst[er];
            atomicAdd(&agg[dn * FV + m],       c0[r]);
            atomicAdd(&agg[dn * FV + m + 16],  c1[r]);
        }
    }
}

// ---------------- out = relu(agg + x @ root + bias) ----------------
__global__ void root_relu_kernel(const float* __restrict__ agg,
                                 const float* __restrict__ xin,
                                 const float* __restrict__ root,  // [32,32]
                                 const float* __restrict__ bias,  // [32]
                                 float* __restrict__ out, int N) {
    int t = blockIdx.x * blockDim.x + threadIdx.x;
    if (t >= N * FV) return;
    int n = t >> 5, o = t & 31;
    float acc = agg[t] + bias[o];
    #pragma unroll
    for (int i = 0; i < FV; ++i) acc += xin[n * FV + i] * root[i * FV + o];
    out[t] = fmaxf(acc, 0.0f);
}

// ---------------- segment max pool (values >= 0 after relu; int-bit atomicMax) ----------------
__global__ void pool_max_kernel(const float* __restrict__ hb,    // [N,32]
                                const int*   __restrict__ batch, // [N]
                                float* __restrict__ pooled,      // [G,64], zero-init
                                int N, int colbase) {
    int t = blockIdx.x * blockDim.x + threadIdx.x;
    if (t >= N * FV) return;
    int n = t >> 5, o = t & 31;
    float v = hb[t];
    int g = batch[n];
    atomicMax((int*)&pooled[g * 64 + colbase + o], __float_as_int(v));
}

// ---------------- head: out = (pooled @ lin0 + b0) @ lin1 + b1 ----------------
__global__ void head_kernel(const float* __restrict__ pooled,   // [G,64]
                            const float* __restrict__ lin0_w,   // [64,64]
                            const float* __restrict__ lin0_b,   // [64]
                            const float* __restrict__ lin1_w,   // [64,1]
                            const float* __restrict__ lin1_b,   // [1]
                            float* __restrict__ out, int G) {
    int g = blockIdx.x * blockDim.x + threadIdx.x;
    if (g >= G) return;
    float acc = lin1_b[0];
    for (int o = 0; o < 64; ++o) {
        float t = lin0_b[o];
        for (int i = 0; i < 64; ++i) t += pooled[g * 64 + i] * lin0_w[i * 64 + o];
        acc += t * lin1_w[o];
    }
    out[g] = acc;
}

// =====================================================================

static inline int ceil_div(int a, int b) { return (a + b - 1) / b; }

extern "C" void kernel_launch(void* const* d_in, const int* in_sizes, int n_in,
                              void* d_out, int out_size, void* d_ws, size_t ws_size,
                              hipStream_t stream) {
    const float* x_p  = (const float*)d_in[0];
    const float* x_d  = (const float*)d_in[1];
    const float* ea_p = (const float*)d_in[2];
    const float* ea_d = (const float*)d_in[3];
    const int*   ei_p = (const int*)d_in[4];
    const int*   ei_d = (const int*)d_in[5];
    const int*   bp   = (const int*)d_in[6];
    const int*   bd   = (const int*)d_in[7];

    const int N = in_sizes[0] / FV;   // 25000
    const int E = in_sizes[2] / FE;   // 100000
    const int G = out_size;           // 512

    // ---- workspace carve (d_ws assumed 256B aligned) ----
    char* ws = (char*)d_ws;
    size_t off = 0;
    auto carve = [&](size_t bytes) {
        void* p = ws + off;
        off = (off + bytes + 255) & ~(size_t)255;
        return p;
    };
    float*    h_buf  = (float*)    carve((size_t)E * HD * 4);
    _Float16* w2t    = (_Float16*) carve((size_t)KR * FV * 2);
    float*    aggb   = (float*)    carve((size_t)N * FV * 4);
    float*    buf0   = (float*)    carve((size_t)N * FV * 4);
    float*    outp   = (float*)    carve((size_t)N * FV * 4);
    float*    outd   = (float*)    carve((size_t)N * FV * 4);
    float*    pooled = (float*)    carve((size_t)G * 64 * 4);
    (void)ws_size; (void)n_in;

    const int BLK = 256;
    const int ntiles = ceil_div(E, 16);

    auto run_layer = [&](const float* xin, const float* ea,
                         const int* src, const int* dst,
                         const float* w1, const float* b1,
                         const float* w2, const float* b2,
                         const float* root, const float* bias,
                         float* out_nodes) {
        edge_mlp_kernel<<<ceil_div(E * HD, BLK), BLK, 0, stream>>>(ea, w1, b1, h_buf, E);
        prep_w2_kernel<<<ceil_div(KR * FV, BLK), BLK, 0, stream>>>(w2, b2, w2t);
        zero_kernel<<<ceil_div(N * FV, BLK), BLK, 0, stream>>>(aggb, N * FV);
        msg_wmma_kernel<<<ceil_div(ntiles * 32, BLK), BLK, 0, stream>>>(
            xin, src, dst, h_buf, w2t, aggb, E, ntiles);
        root_relu_kernel<<<ceil_div(N * FV, BLK), BLK, 0, stream>>>(
            aggb, xin, root, bias, out_nodes, N);
    };

    auto run_branch = [&](const float* x0, const float* ea, const int* ei,
                          int pb, float* out_nodes) {
        const int* src = ei;        // edge_index[0]
        const int* dst = ei + E;    // edge_index[1]
        const float* w1_0 = (const float*)d_in[pb + 0];
        const float* b1_0 = (const float*)d_in[pb + 1];
        const float* w2_0 = (const float*)d_in[pb + 2];
        const float* b2_0 = (const float*)d_in[pb + 3];
        const float* rt_0 = (const float*)d_in[pb + 4];
        const float* bi_0 = (const float*)d_in[pb + 5];
        const float* w1_1 = (const float*)d_in[pb + 6];
        const float* b1_1 = (const float*)d_in[pb + 7];
        const float* w2_1 = (const float*)d_in[pb + 8];
        const float* b2_1 = (const float*)d_in[pb + 9];
        const float* rt_1 = (const float*)d_in[pb + 10];
        const float* bi_1 = (const float*)d_in[pb + 11];
        run_layer(x0,   ea, src, dst, w1_0, b1_0, w2_0, b2_0, rt_0, bi_0, buf0);
        run_layer(buf0, ea, src, dst, w1_1, b1_1, w2_1, b2_1, rt_1, bi_1, out_nodes);
    };

    run_branch(x_p, ea_p, ei_p, 8,  outp);
    run_branch(x_d, ea_d, ei_d, 20, outd);

    zero_kernel<<<ceil_div(G * 64, BLK), BLK, 0, stream>>>(pooled, G * 64);
    pool_max_kernel<<<ceil_div(N * FV, BLK), BLK, 0, stream>>>(outp, bp, pooled, N, 0);
    pool_max_kernel<<<ceil_div(N * FV, BLK), BLK, 0, stream>>>(outd, bd, pooled, N, 32);

    const float* lin0_w = (const float*)d_in[32];
    const float* lin0_b = (const float*)d_in[33];
    const float* lin1_w = (const float*)d_in[34];
    const float* lin1_b = (const float*)d_in[35];
    head_kernel<<<ceil_div(G, 64), 64, 0, stream>>>(
        pooled, lin0_w, lin0_b, lin1_w, lin1_b, (float*)d_out, G);
}
